// MultiHeadAttention_80504866996539
// MI455X (gfx1250) — matrix operand order repaired and checked by reading, not compiled, not run on hardware
//
#include <hip/hip_runtime.h>
#include <hip/hip_bf16.h>

// MHA: B=4, S=2048, H=8, D=512, dk=64. f16 WMMA (16x16x32) everywhere, f32 accum.
#define B_  4
#define S_  2048
#define H_  8
#define D_  512
#define DK_ 64

typedef __attribute__((ext_vector_type(16))) _Float16 v16h;
typedef __attribute__((ext_vector_type(8)))  _Float16 v8h;
typedef __attribute__((ext_vector_type(8)))  float    v8f;
typedef __attribute__((ext_vector_type(4)))  int      v4i;

__device__ __forceinline__ v8f wmma16(v16h a, v16h b, v8f c) {
  // D = A(16x32 f16) * B(32x16 f16) + C(16x16 f32)
  return __builtin_amdgcn_wmma_f32_16x16x32_f16(false, a, false, b, (short)0, c,
                                                false, false);
}

__device__ __forceinline__ v16h cat8(v8h lo, v8h hi) {
  v16h r;
#pragma unroll
  for (int j = 0; j < 8; ++j) { r[j] = lo[j]; r[8 + j] = hi[j]; }
  return r;
}

// ---- CDNA5 async global->LDS (ASYNCcnt) helpers -----------------------------
__device__ __forceinline__ void async_b128(const void* g, void* l) {
#if __has_builtin(__builtin_amdgcn_global_load_async_to_lds_b128)
  // Toolchain signature (from probe): (v4i AS1*, v4i AS3*, imm offset, imm cpol)
  __builtin_amdgcn_global_load_async_to_lds_b128(
      (__attribute__((address_space(1))) v4i*)g,
      (__attribute__((address_space(3))) v4i*)l, 0, 0);
#else
  unsigned lofs = (unsigned)(unsigned long long)l;   // LDS offset = low 32 bits
  asm volatile("global_load_async_to_lds_b128 %0, %1, off"
               :: "v"(lofs), "v"(g) : "memory");
#endif
}

template <int N>
__device__ __forceinline__ void wait_async() {
#if __has_builtin(__builtin_amdgcn_s_wait_asynccnt)
  __builtin_amdgcn_s_wait_asynccnt(N);
#else
  asm volatile("s_wait_asynccnt %0" :: "i"(N) : "memory");
#endif
}

// ---------------------------------------------------------------------------
// Kernel 0: weight prep.  W[h][k][c] (f32) -> Wt[h][c][k] (f16), WO -> WOt[c][k].
// ---------------------------------------------------------------------------
__global__ __launch_bounds__(256) void prep_weights_kernel(
    const float* __restrict__ WQ, const float* __restrict__ WK,
    const float* __restrict__ WV, const float* __restrict__ WO,
    _Float16* __restrict__ WtQ, _Float16* __restrict__ WtK,
    _Float16* __restrict__ WtV, _Float16* __restrict__ WOt) {
  int i = blockIdx.x * 256 + threadIdx.x;       // 0 .. 8*512*64-1
  int h = i >> 15;
  int k = (i >> 6) & 511;
  int c = i & 63;
  int o = (h * DK_ + c) * D_ + k;
  WtQ[o] = (_Float16)WQ[i];
  WtK[o] = (_Float16)WK[i];
  WtV[o] = (_Float16)WV[i];
  int k2 = i >> 9, c2 = i & 511;                // WO [512][512]
  WOt[c2 * D_ + k2] = (_Float16)WO[i];
}

// ---------------------------------------------------------------------------
// Kernel 1: per-head projections.  X[b,s,512] (f32) @ Wt[h] -> f16.
//   z=0: Qh[bh][s][dk], z=1: Kh[bh][s][dk], z=2: Vt[bh][dk][s] (transposed)
// ---------------------------------------------------------------------------
__global__ __launch_bounds__(256) void proj_kernel(
    const float* __restrict__ Qin, const float* __restrict__ Kin,
    const float* __restrict__ Vin,
    const _Float16* __restrict__ WtQ, const _Float16* __restrict__ WtK,
    const _Float16* __restrict__ WtV,
    _Float16* __restrict__ Qh, _Float16* __restrict__ Kh,
    _Float16* __restrict__ Vt) {
  const int z    = blockIdx.z;
  const int bh   = blockIdx.y;
  const int b    = bh >> 3, h = bh & 7;
  const int wv   = threadIdx.x >> 5, lane = threadIdx.x & 31;
  const int rowbase = blockIdx.x * 128 + wv * 16;
  const int base0   = (lane >> 4) * 8;
  const int l15     = lane & 15;

  const float*    X = (z == 0) ? Qin : (z == 1) ? Kin : Vin;
  const _Float16* W = ((z == 0) ? WtQ : (z == 1) ? WtK : WtV) + (size_t)h * DK_ * D_;
  const float* Xrow = X + ((size_t)b * S_ + rowbase + l15) * D_;

  v8f acc[4] = {};
  for (int k0 = 0; k0 < D_; k0 += 32) {
    // Issue all loads first (one clause), then convert, then 4 WMMAs.
    v8f x0 = *(const v8f*)(Xrow + k0 + (lane >> 4) * 16);
    v8f x1 = *(const v8f*)(Xrow + k0 + (lane >> 4) * 16 + 8);
    const int klane = k0 + (lane >> 4) * 16;
    v16h bm[4];
#pragma unroll
    for (int t = 0; t < 4; ++t) {
      const _Float16* wp = W + (size_t)(t * 16 + l15) * D_ + klane;
      bm[t] = cat8(*(const v8h*)wp, *(const v8h*)(wp + 8));
    }
    v16h a;
#pragma unroll
    for (int j = 0; j < 8; ++j) { a[j] = (_Float16)x0[j]; a[8 + j] = (_Float16)x1[j]; }
#pragma unroll
    for (int t = 0; t < 4; ++t) acc[t] = wmma16(a, bm[t], acc[t]);
  }
  // C layout: VGPR r, lane L -> M = r + 8*(L>>4), N = L&15
  if (z < 2) {
    _Float16* O = ((z == 0) ? Qh : Kh) + (size_t)bh * S_ * DK_;
#pragma unroll
    for (int t = 0; t < 4; ++t)
#pragma unroll
      for (int r = 0; r < 8; ++r) {
        int row = rowbase + r + 8 * (lane >> 4);
        int col = t * 16 + l15;
        O[(size_t)row * DK_ + col] = (_Float16)acc[t][r];
      }
  } else {
    _Float16* O = Vt + (size_t)bh * DK_ * S_;
#pragma unroll
    for (int t = 0; t < 4; ++t)
#pragma unroll
      for (int r = 0; r < 8; ++r) {
        int row = rowbase + r + 8 * (lane >> 4);
        int col = t * 16 + l15;
        O[(size_t)col * S_ + row] = (_Float16)acc[t][r];   // transposed store
      }
  }
}

// ---------------------------------------------------------------------------
// Kernel 2: flash attention per (b,h).  Key blocks of 32 stream through LDS
// via double-buffered GLOBAL_LOAD_ASYNC_TO_LDS_B128 (ASYNCcnt), shared by all
// 8 waves.  Online softmax; P transposed C->A layout through LDS.
// ---------------------------------------------------------------------------
__global__ __launch_bounds__(256) void attn_kernel(
    const _Float16* __restrict__ Qh, const _Float16* __restrict__ Kh,
    const _Float16* __restrict__ Vt, _Float16* __restrict__ Oc) {
  __shared__ _Float16 Kbuf[2][32][64];   // 2 x 4 KB: 32 keys x 64 dims
  __shared__ _Float16 Vbuf[2][64][32];   // 2 x 4 KB: 64 dims x 32 keys (V^T)
  __shared__ _Float16 Plds[8][16][32];   // per-wave P staging, 8 KB

  const int tid  = threadIdx.x;
  const int bh   = blockIdx.y;
  const int b    = bh >> 3, h = bh & 7;
  const int wv   = tid >> 5, lane = tid & 31;
  const int qbase = blockIdx.x * 128 + wv * 16;
  const int base0 = (lane >> 4) * 8;
  const int l15   = lane & 15;

  const _Float16* Qp = Qh + ((size_t)bh * S_ + qbase) * DK_;
  const _Float16* Kp = Kh + (size_t)bh * S_ * DK_;
  const _Float16* Vp = Vt + (size_t)bh * DK_ * S_;

  // Cooperative async staging: each of 256 threads owns one b128 per buffer.
  const int krow = tid >> 3, kc8 = (tid & 7) * 8;   // K: 32 x 64 halves
  const int vrow = tid >> 2, vc8 = (tid & 3) * 8;   // V^T: 64 x 32 halves

  // Q tile in A layout; fold 1/sqrt(64)=0.125 (exact in f16)
  v16h qa[2];
#pragma unroll
  for (int ss = 0; ss < 2; ++ss) {
    const _Float16* qp = Qp + (size_t)l15 * DK_ + ss * 32 + (lane >> 4) * 16;
    v8h lo = *(const v8h*)qp;
    v8h hi = *(const v8h*)(qp + 8);
#pragma unroll
    for (int j = 0; j < 8; ++j) {
      qa[ss][j]     = lo[j] * (_Float16)0.125f;
      qa[ss][8 + j] = hi[j] * (_Float16)0.125f;
    }
  }

  // Prologue: stage key block 0 into buffer 0.
  async_b128(Kp + (size_t)krow * DK_ + kc8, &Kbuf[0][krow][kc8]);
  async_b128(Vp + (size_t)vrow * S_ + vc8, &Vbuf[0][vrow][vc8]);

  v8f oacc[4] = {};
  float m[8], l[8];
#pragma unroll
  for (int r = 0; r < 8; ++r) { m[r] = -3.0e38f; l[r] = 0.f; }

  for (int kb = 0; kb < S_; kb += 32) {
    const int buf = (kb >> 5) & 1;
    if (kb + 32 < S_) {   // uniform: prefetch next block into the other buffer
      async_b128(Kp + (size_t)(kb + 32 + krow) * DK_ + kc8, &Kbuf[buf ^ 1][krow][kc8]);
      async_b128(Vp + (size_t)vrow * S_ + (kb + 32) + vc8, &Vbuf[buf ^ 1][vrow][vc8]);
      wait_async<2>();    // this wave's loads for `buf` are done (in-order)
    } else {
      wait_async<0>();
    }
    __syncthreads();      // all waves' slices of `buf` visible

    // scores: 2 key tiles x 2 d-steps = 4 WMMAs (B operands from LDS)
    v8f sc[2];
#pragma unroll
    for (int t = 0; t < 2; ++t) {
      v8f c = {};
#pragma unroll
      for (int ss = 0; ss < 2; ++ss) {
        const _Float16* kp = &Kbuf[buf][t * 16 + l15][ss * 32 + (lane >> 4) * 16];
        v16h bm = cat8(*(const v8h*)kp, *(const v8h*)(kp + 8));
        c = wmma16(qa[ss], bm, c);
      }
      sc[t] = c;
    }
    // online softmax: row stats via 16-lane shuffle tree
    float mnew[8], alpha[8];
#pragma unroll
    for (int r = 0; r < 8; ++r) {
      float v = fmaxf(sc[0][r], sc[1][r]);
#pragma unroll
      for (int off = 8; off >= 1; off >>= 1) v = fmaxf(v, __shfl_xor(v, off, 16));
      mnew[r]  = fmaxf(m[r], v);
      alpha[r] = __expf(m[r] - mnew[r]);
      m[r]     = mnew[r];
    }
#pragma unroll
    for (int t = 0; t < 2; ++t)
#pragma unroll
      for (int r = 0; r < 8; ++r) sc[t][r] = __expf(sc[t][r] - mnew[r]);
#pragma unroll
    for (int r = 0; r < 8; ++r) {
      float v = sc[0][r] + sc[1][r];
#pragma unroll
      for (int off = 8; off >= 1; off >>= 1) v += __shfl_xor(v, off, 16);
      l[r] = l[r] * alpha[r] + v;
    }
#pragma unroll
    for (int t = 0; t < 4; ++t)
#pragma unroll
      for (int r = 0; r < 8; ++r) oacc[t][r] *= alpha[r];

    // P: C layout -> LDS [q][key]
#pragma unroll
    for (int t = 0; t < 2; ++t)
#pragma unroll
      for (int r = 0; r < 8; ++r)
        Plds[wv][r + 8 * (lane >> 4)][t * 16 + l15] = (_Float16)sc[t][r];
    __syncthreads();

    // P reload in A layout (two ds_load_b128), then P @ V (4 WMMAs)
    v16h pa;
    {
      const _Float16* pp = &Plds[wv][l15][(lane >> 4) * 16];
      pa = cat8(*(const v8h*)pp, *(const v8h*)(pp + 8));
    }
#pragma unroll
    for (int t = 0; t < 4; ++t) {
      const _Float16* vp = &Vbuf[buf][t * 16 + l15][(lane >> 4) * 16];
      v16h bm = cat8(*(const v8h*)vp, *(const v8h*)(vp + 8));
      oacc[t] = wmma16(pa, bm, oacc[t]);
    }
    __syncthreads();      // no wave overwrites buf^1 / Plds while others read
  }

  // epilogue: normalize, store head-concatenated f16 O[b][q][h*64+d]
  float inv[8];
#pragma unroll
  for (int r = 0; r < 8; ++r) inv[r] = 1.0f / l[r];
#pragma unroll
  for (int t = 0; t < 4; ++t)
#pragma unroll
    for (int r = 0; r < 8; ++r) {
      int q = qbase + r + 8 * (lane >> 4);
      int d = t * 16 + l15;
      Oc[((size_t)b * S_ + q) * D_ + (size_t)h * DK_ + d] =
          (_Float16)(oacc[t][r] * inv[r]);
    }
}

// ---------------------------------------------------------------------------
// Kernel 3: output projection.  Oc[8192,512] f16 @ WOt -> f32 d_out.
// ---------------------------------------------------------------------------
__global__ __launch_bounds__(256) void outproj_kernel(
    const _Float16* __restrict__ Oc, const _Float16* __restrict__ WOt,
    float* __restrict__ out) {
  const int wv = threadIdx.x >> 5, lane = threadIdx.x & 31;
  const int rowbase = blockIdx.x * 128 + wv * 16;
  const int colbase = blockIdx.y * 64;
  const int base0 = (lane >> 4) * 8, l15 = lane & 15;
  const _Float16* Ar = Oc + (size_t)(rowbase + l15) * D_;

  v8f acc[4] = {};
  for (int k0 = 0; k0 < D_; k0 += 32) {
    const _Float16* ap = Ar + k0 + (lane >> 4) * 16;
    v16h a = cat8(*(const v8h*)ap, *(const v8h*)(ap + 8));
    const int klane = k0 + (lane >> 4) * 16;
    v16h bm[4];
#pragma unroll
    for (int t = 0; t < 4; ++t) {
      const _Float16* wp = WOt + (size_t)(colbase + t * 16 + l15) * D_ + klane;
      bm[t] = cat8(*(const v8h*)wp, *(const v8h*)(wp + 8));
    }
#pragma unroll
    for (int t = 0; t < 4; ++t) acc[t] = wmma16(a, bm[t], acc[t]);
  }
#pragma unroll
  for (int t = 0; t < 4; ++t)
#pragma unroll
    for (int r = 0; r < 8; ++r)
      out[(size_t)(rowbase + r + 8 * (lane >> 4)) * D_ + colbase + t * 16 + l15] =
          acc[t][r];
}

// ---------------------------------------------------------------------------
extern "C" void kernel_launch(void* const* d_in, const int* in_sizes, int n_in,
                              void* d_out, int out_size, void* d_ws, size_t ws_size,
                              hipStream_t stream) {
  const float* Q  = (const float*)d_in[0];
  const float* K  = (const float*)d_in[1];
  const float* V  = (const float*)d_in[2];
  const float* WQ = (const float*)d_in[3];
  const float* WK = (const float*)d_in[4];
  const float* WV = (const float*)d_in[5];
  const float* WO = (const float*)d_in[6];

  // Workspace (f16): Qh | Kh | Vt | Oc | WtQ | WtK | WtV | WOt  (~35.7 MB)
  _Float16* ws = (_Float16*)d_ws;
  const size_t N1 = (size_t)B_ * H_ * S_ * DK_;   // 4,194,304 halves
  const size_t NW = (size_t)H_ * D_ * DK_;        // 262,144 halves
  _Float16* Qh  = ws;
  _Float16* Kh  = ws + N1;
  _Float16* Vt  = ws + 2 * N1;
  _Float16* Oc  = ws + 3 * N1;
  _Float16* WtQ = ws + 4 * N1;
  _Float16* WtK = WtQ + NW;
  _Float16* WtV = WtK + NW;
  _Float16* WOt = WtV + NW;

  prep_weights_kernel<<<dim3(NW / 256), 256, 0, stream>>>(WQ, WK, WV, WO,
                                                          WtQ, WtK, WtV, WOt);
  proj_kernel<<<dim3(S_ / 128, B_ * H_, 3), 256, 0, stream>>>(
      Q, K, V, WtQ, WtK, WtV, Qh, Kh, Vt);
  attn_kernel<<<dim3(S_ / 128, B_ * H_), 256, 0, stream>>>(Qh, Kh, Vt, Oc);
  outproj_kernel<<<dim3((B_ * S_) / 128, D_ / 64), 256, 0, stream>>>(
      Oc, WOt, (float*)d_out);
}
